// Local_Correlation_37014028157089
// MI455X (gfx1250) — matrix operand order repaired and checked by reading, not compiled, not run on hardware
//
#include <hip/hip_runtime.h>
#include <hip/hip_bf16.h>

// Local correlation / non-local attention block on gfx1250 (MI455X).
// All GEMM stages run on v_wmma_f32_16x16x32_f16 (wave32 WMMA).
//
// b=2, c=inter=64, n=16384, k=20 (padded to 32 for 16x16 tiles).

typedef _Float16 v16h __attribute__((ext_vector_type(16)));
typedef _Float16 v2h  __attribute__((ext_vector_type(2)));
typedef float    v8f  __attribute__((ext_vector_type(8)));

#define NPOINTS_TOTAL 32768   // b*n
#define NDIM 16384
#define CDIM 64
#define KNEI 20
#define NPTS 4                // points per wave per kernel call

__device__ __forceinline__ v8f wmma_f16(v16h a, v16h b, v8f c) {
    return __builtin_amdgcn_wmma_f32_16x16x32_f16(
        /*neg_a=*/false, a, /*neg_b=*/false, b,
        /*c_mod=*/(short)0, c, /*reuse_a=*/false, /*reuse_b=*/false);
}

// A-matrix 16x32 f16 tile (ISA 7.12.2): lane lm = M row; lanes<16: VGPR j<4 -> K=2j,
// VGPR j>=4 -> K=16+2(j-4); lanes>=16: +8 / +24. buf is row-major [M][K], row stride S.
__device__ __forceinline__ v16h load_a_tile(const _Float16* buf, int S,
                                            int mbase, int kbase, int lm, bool hi) {
    v16h a;
    const _Float16* row = buf + (mbase + lm) * S + kbase;
    const int o1 = hi ? 8 : 0;
    const int o2 = hi ? 24 : 16;
#pragma unroll
    for (int j = 0; j < 4; ++j) {
        a[2 * j]     = row[o1 + 2 * j];
        a[2 * j + 1] = row[o1 + 2 * j + 1];
    }
#pragma unroll
    for (int j = 0; j < 4; ++j) {
        a[8 + 2 * j] = row[o2 + 2 * j];
        a[9 + 2 * j] = row[o2 + 2 * j + 1];
    }
    return a;
}

// B-matrix 32x16 f16 tile: same K order as A: lanes<16: VGPR j<4 -> K=2j, VGPR j>=4 ->
// K=16+2(j-4); lanes>=16: +8 / +24. N = lane%16. buf holds B[k][n] as buf[n*S + k] (K contiguous).
__device__ __forceinline__ v16h load_b_tile(const _Float16* buf, int S,
                                            int kbase, int nbase, int lm, bool hi) {
    v16h b;
    const _Float16* row = buf + (nbase + lm) * S + kbase + (hi ? 8 : 0);
#pragma unroll
    for (int j = 0; j < 8; ++j) {
        b[2 * j]     = row[(j < 4 ? 0 : 8) + 2 * j];
        b[2 * j + 1] = row[(j < 4 ? 0 : 8) + 2 * j + 1];
    }
    return b;
}

// ---- prep: transpose x (b,c,n) f32 -> xt[(b*n)][c] f16 (coalesced gather rows) ----
__global__ __launch_bounds__(256) void lc_prep_x(const float* __restrict__ x,
                                                 _Float16* __restrict__ xt) {
    int tid = blockIdx.x * 256 + threadIdx.x;      // == (bI*64 + c)*16384 + nI
    int nI = tid & (NDIM - 1);
    int c  = (tid >> 14) & 63;
    int bI = tid >> 20;
    xt[(((bI << 14) | nI) << 6) + c] = (_Float16)x[tid];
}

// ---- prep: Wg,Wt,Wp,Ww (64x64 f32, row-major [o][c]) -> f16 concat in ws ----
__global__ __launch_bounds__(256) void lc_prep_w(const float* __restrict__ Wg,
                                                 const float* __restrict__ Wt,
                                                 const float* __restrict__ Wp,
                                                 const float* __restrict__ Ww,
                                                 _Float16* __restrict__ wout) {
    int tid = blockIdx.x * 256 + threadIdx.x;      // 0..16383
    int which = tid >> 12;
    int off   = tid & 4095;
    const float* src = (which == 0) ? Wg : (which == 1) ? Wt : (which == 2) ? Wp : Ww;
    wout[tid] = (_Float16)src[off];
}

// ---- main: 2 waves/block, each wave handles NPTS points end-to-end ----
__global__ __launch_bounds__(64, 4) void lc_main(
    const _Float16* __restrict__ xt,   // [b*n][64] f16
    const _Float16* __restrict__ wts,  // Wg|Wt|Wp|Ww, each [64][64] f16
    const int* __restrict__ idx,       // [b*n][20]
    const float* __restrict__ bg, const float* __restrict__ bt,
    const float* __restrict__ bp, const float* __restrict__ bw,
    const float* __restrict__ gmm, const float* __restrict__ bta,
    float* __restrict__ out) {
    __shared__ _Float16 sE[2][32 * 64];   // edge feats e, later reused for y
    __shared__ _Float16 sTh[2][32 * 64];  // theta, row-major [k][c]
    __shared__ _Float16 sPh[2][32 * 64];  // phi,   row-major [k][c]
    __shared__ _Float16 sGt[2][64 * 32];  // g transposed [c][k]
    __shared__ float    sF[2][32 * 32];   // attention logits f32
    __shared__ _Float16 sAt[2][32 * 32];  // softmaxed attn f16

    // wave id is uniform within the wave: tell the compiler so pt-derived
    // addresses (idx loads) become scalar (SMEM) loads.
    const int  w    = __builtin_amdgcn_readfirstlane((int)(threadIdx.x >> 5));
    const int  lane = threadIdx.x & 31;
    const int  lm   = lane & 15;
    const bool hi   = lane >= 16;
    const int  rofs = hi ? 8 : 0;        // C-tile row offset for this lane half
    const int  gw   = blockIdx.x * 2 + w;

    _Float16* eb = sE[w];
    _Float16* th = sTh[w];
    _Float16* ph = sPh[w];
    _Float16* gt = sGt[w];
    float*    fb = sF[w];
    _Float16* ab = sAt[w];

    const _Float16* Wg16 = wts;
    const _Float16* Wt16 = wts + 4096;
    const _Float16* Wp16 = wts + 8192;
    const _Float16* Ww16 = wts + 12288;
    const float INVS = 0.99999500003749978f;  // 1/sqrt(1+1e-5)
    const v8f z8 = {0.f, 0.f, 0.f, 0.f, 0.f, 0.f, 0.f, 0.f};

#pragma unroll 1
    for (int i = 0; i < NPTS; ++i) {
        const int pt = gw * NPTS + i;
        const int bI = pt >> 14;
        const int nI = pt & (NDIM - 1);

        // ---- stage e = neighbor - center into LDS (f16, rows 20..31 zero) ----
        const unsigned* x32 = (const unsigned*)xt;   // 32 dwords per feature row
        unsigned cw = x32[pt * 32 + lane];
        v2h ch;
        { union { unsigned u; v2h h; } u; u.u = cw; ch = u.h; }
        unsigned* e32 = (unsigned*)eb;
#pragma unroll 4
        for (int kk = 0; kk < KNEI; ++kk) {
            int gi = idx[pt * KNEI + kk];            // wave-uniform -> s_load
            unsigned nw = x32[gi * 32 + lane];       // coalesced 128B row
            union { unsigned u; v2h h; } un, ud;
            un.u = nw;
            ud.h = un.h - ch;
            e32[kk * 32 + lane] = ud.u;
        }
#pragma unroll
        for (int kk = KNEI; kk < 32; ++kk) e32[kk * 32 + lane] = 0u;
        __syncthreads();

        // ---- projections: theta/phi/g = e @ W^T + b ----
        v16h Ae[2][2];
#pragma unroll
        for (int mt = 0; mt < 2; ++mt)
#pragma unroll
            for (int kc = 0; kc < 2; ++kc)
                Ae[mt][kc] = load_a_tile(eb, 64, mt * 16, kc * 32, lm, hi);

        for (int nt = 0; nt < 4; ++nt) {
            const int ncol = nt * 16 + lm;
            {   // theta -> th row-major; bias folded into C init
                v16h B0 = load_b_tile(Wt16, 64, 0, nt * 16, lm, hi);
                v16h B1 = load_b_tile(Wt16, 64, 32, nt * 16, lm, hi);
                const float bv = bt[ncol];
                v8f a0, a1;
#pragma unroll
                for (int r = 0; r < 8; ++r) { a0[r] = bv; a1[r] = bv; }
                a0 = wmma_f16(Ae[0][0], B0, a0);      // independent chains
                a1 = wmma_f16(Ae[1][0], B0, a1);
                a0 = wmma_f16(Ae[0][1], B1, a0);
                a1 = wmma_f16(Ae[1][1], B1, a1);
#pragma unroll
                for (int r = 0; r < 8; ++r) {
                    th[(rofs + r) * 64 + ncol]        = (_Float16)a0[r];
                    th[(16 + rofs + r) * 64 + ncol]   = (_Float16)a1[r];
                }
            }
            {   // phi -> ph row-major
                v16h B0 = load_b_tile(Wp16, 64, 0, nt * 16, lm, hi);
                v16h B1 = load_b_tile(Wp16, 64, 32, nt * 16, lm, hi);
                const float bv = bp[ncol];
                v8f a0, a1;
#pragma unroll
                for (int r = 0; r < 8; ++r) { a0[r] = bv; a1[r] = bv; }
                a0 = wmma_f16(Ae[0][0], B0, a0);
                a1 = wmma_f16(Ae[1][0], B0, a1);
                a0 = wmma_f16(Ae[0][1], B1, a0);
                a1 = wmma_f16(Ae[1][1], B1, a1);
#pragma unroll
                for (int r = 0; r < 8; ++r) {
                    ph[(rofs + r) * 64 + ncol]        = (_Float16)a0[r];
                    ph[(16 + rofs + r) * 64 + ncol]   = (_Float16)a1[r];
                }
            }
            {   // g -> gt transposed [c][k]
                v16h B0 = load_b_tile(Wg16, 64, 0, nt * 16, lm, hi);
                v16h B1 = load_b_tile(Wg16, 64, 32, nt * 16, lm, hi);
                const float bv = bg[ncol];
                v8f a0, a1;
#pragma unroll
                for (int r = 0; r < 8; ++r) { a0[r] = bv; a1[r] = bv; }
                a0 = wmma_f16(Ae[0][0], B0, a0);
                a1 = wmma_f16(Ae[1][0], B0, a1);
                a0 = wmma_f16(Ae[0][1], B1, a0);
                a1 = wmma_f16(Ae[1][1], B1, a1);
#pragma unroll
                for (int r = 0; r < 8; ++r) {
                    gt[ncol * 32 + rofs + r]          = (_Float16)a0[r];
                    gt[ncol * 32 + 16 + rofs + r]     = (_Float16)a1[r];
                }
            }
        }
        __syncthreads();

        // ---- f = theta @ phi^T : 4 independent accumulator chains ----
        {
            v8f f00 = z8, f01 = z8, f10 = z8, f11 = z8;
#pragma unroll
            for (int kc = 0; kc < 2; ++kc) {
                v16h A0 = load_a_tile(th, 64, 0, kc * 32, lm, hi);
                v16h A1 = load_a_tile(th, 64, 16, kc * 32, lm, hi);
                v16h B0 = load_b_tile(ph, 64, kc * 32, 0, lm, hi);
                v16h B1 = load_b_tile(ph, 64, kc * 32, 16, lm, hi);
                f00 = wmma_f16(A0, B0, f00);
                f10 = wmma_f16(A1, B0, f10);
                f01 = wmma_f16(A0, B1, f01);
                f11 = wmma_f16(A1, B1, f11);
            }
#pragma unroll
            for (int r = 0; r < 8; ++r) {
                fb[(rofs + r) * 32 + lm]           = f00[r];
                fb[(rofs + r) * 32 + 16 + lm]      = f01[r];
                fb[(16 + rofs + r) * 32 + lm]      = f10[r];
                fb[(16 + rofs + r) * 32 + 16 + lm] = f11[r];
            }
        }
        __syncthreads();

        // ---- row softmax over j<20; attn padded with zeros ----
        {
            const float* frow = fb + lane * 32;
            _Float16* arow = ab + lane * 32;
            if (lane < KNEI) {
                float mmax = -3.0e38f;
#pragma unroll
                for (int j = 0; j < KNEI; ++j) mmax = fmaxf(mmax, frow[j]);
                float ev[KNEI];
                float s = 0.f;
#pragma unroll
                for (int j = 0; j < KNEI; ++j) { ev[j] = __expf(frow[j] - mmax); s += ev[j]; }
                float rs = 1.0f / s;
#pragma unroll
                for (int j = 0; j < KNEI; ++j) arow[j] = (_Float16)(ev[j] * rs);
#pragma unroll
                for (int j = KNEI; j < 32; ++j) arow[j] = (_Float16)0.f;
            } else {
#pragma unroll
                for (int j = 0; j < 32; ++j) arow[j] = (_Float16)0.f;
            }
        }
        __syncthreads();

        // ---- y = attn @ g  (reuse eb as y, row-major [k][c]) ----
        {
            v16h Aat0 = load_a_tile(ab, 32, 0, 0, lm, hi);
            v16h Aat1 = load_a_tile(ab, 32, 16, 0, lm, hi);
#pragma unroll
            for (int nt = 0; nt < 4; ++nt) {
                v16h B = load_b_tile(gt, 32, 0, nt * 16, lm, hi);
                v8f a0 = wmma_f16(Aat0, B, z8);
                v8f a1 = wmma_f16(Aat1, B, z8);
#pragma unroll
                for (int r = 0; r < 8; ++r) {
                    eb[(rofs + r) * 64 + nt * 16 + lm]      = (_Float16)a0[r];
                    eb[(16 + rofs + r) * 64 + nt * 16 + lm] = (_Float16)a1[r];
                }
            }
        }
        __syncthreads();

        // ---- z = y @ Ww^T, fused BN affine, max over valid rows, store ----
        v16h Ay[2][2];
#pragma unroll
        for (int mt = 0; mt < 2; ++mt)
#pragma unroll
            for (int kc = 0; kc < 2; ++kc)
                Ay[mt][kc] = load_a_tile(eb, 64, mt * 16, kc * 32, lm, hi);

        for (int nt = 0; nt < 4; ++nt) {
            v16h B0 = load_b_tile(Ww16, 64, 0, nt * 16, lm, hi);
            v16h B1 = load_b_tile(Ww16, 64, 32, nt * 16, lm, hi);
            const int p = nt * 16 + lm;
            const float sgp = INVS * gmm[p];            // z*sgp + cc
            const float cc  = bw[p] * sgp + bta[p];
            v8f a0 = z8, a1 = z8;
            a0 = wmma_f16(Ay[0][0], B0, a0);
            a1 = wmma_f16(Ay[1][0], B0, a1);
            a0 = wmma_f16(Ay[0][1], B1, a0);
            a1 = wmma_f16(Ay[1][1], B1, a1);
            float mx = -3.0e38f;
#pragma unroll
            for (int r = 0; r < 8; ++r)                 // rows 0..15: all valid
                mx = fmaxf(mx, a0[r] * sgp + cc);
            if (!hi) {                                  // rows 16..19 only
#pragma unroll
                for (int r = 0; r < 4; ++r)
                    mx = fmaxf(mx, a1[r] * sgp + cc);
            }
            mx = fmaxf(mx, __shfl_xor(mx, 16, 32));     // combine M halves
            if (!hi) out[(bI * CDIM + p) * NDIM + nI] = mx;
        }
        __syncthreads();  // protect LDS reuse before next point
    }
}

extern "C" void kernel_launch(void* const* d_in, const int* in_sizes, int n_in,
                              void* d_out, int out_size, void* d_ws, size_t ws_size,
                              hipStream_t stream) {
    const float* x    = (const float*)d_in[0];
    const int*   idx  = (const int*)d_in[1];
    const float* Wg   = (const float*)d_in[2];
    const float* bg   = (const float*)d_in[3];
    const float* Wt   = (const float*)d_in[4];
    const float* bt   = (const float*)d_in[5];
    const float* Wp   = (const float*)d_in[6];
    const float* bp   = (const float*)d_in[7];
    const float* Ww   = (const float*)d_in[8];
    const float* bw   = (const float*)d_in[9];
    const float* gmm  = (const float*)d_in[10];
    const float* bta  = (const float*)d_in[11];
    float* out = (float*)d_out;

    _Float16* xt16 = (_Float16*)d_ws;               // 2*16384*64 f16 = 4 MB
    _Float16* w16  = xt16 + (size_t)NPOINTS_TOTAL * CDIM;  // 4*4096 f16 = 32 KB

    lc_prep_x<<<(NPOINTS_TOTAL * CDIM) / 256, 256, 0, stream>>>(x, xt16);
    lc_prep_w<<<(4 * 4096) / 256, 256, 0, stream>>>(Wg, Wt, Wp, Ww, w16);

    // 32768 points / (2 waves * NPTS points) = 4096 blocks of 64 threads
    lc_main<<<NPOINTS_TOTAL / (2 * NPTS), 64, 0, stream>>>(
        xt16, w16, idx, bg, bt, bp, bw, gmm, bta, out);
}
